// MultiHeadSelfAttention_103079215446
// MI455X (gfx1250) — matrix operand order repaired and checked by reading, not compiled, not run on hardware
//
#include <hip/hip_runtime.h>
#include <hip/hip_bf16.h>

typedef __attribute__((ext_vector_type(16))) __bf16 v16bf;
typedef __attribute__((ext_vector_type(8)))  float  v8f;
typedef unsigned short u16;
typedef unsigned int   u32;

union AFrag { v16bf v; uint4 q[2]; u16 h[16]; };

#if defined(__gfx1250__)
#define HAVE_ASYNC_LDS 1
#else
#define HAVE_ASYNC_LDS 0
#endif

// Copy 16 bytes global -> LDS.
// gfx1250: GLOBAL_LOAD_ASYNC_TO_LDS_B128 (ASYNCcnt-tracked, no VGPR data).
// VDST = LDS byte offset (low 32 bits of the generic shared pointer: the LDS
// aperture keeps the offset in addr[31:0]); VADDR = 64-bit global address.
__device__ __forceinline__ void cp_b128(void* lds_dst, const void* gsrc) {
#if HAVE_ASYNC_LDS
  u32 loff = (u32)(unsigned long long)lds_dst;
  asm volatile("global_load_async_to_lds_b128 %0, %1, off"
               :: "v"(loff), "v"(gsrc) : "memory");
#else
  *(uint4*)lds_dst = *(const uint4*)gsrc;
#endif
}

// Wait until at most N async global<->LDS ops remain outstanding.
__device__ __forceinline__ void wait_async_le8() {
#if HAVE_ASYNC_LDS
  asm volatile("s_wait_asynccnt 0x8" ::: "memory");
#endif
}
__device__ __forceinline__ void wait_async_0() {
#if HAVE_ASYNC_LDS
  asm volatile("s_wait_asynccnt 0x0" ::: "memory");
#endif
}

__device__ __forceinline__ u16 f32_to_bf16_rne(float f) {
  union { float f; u32 u; } cv; cv.f = f;
  u32 u = cv.u;
  u32 r = u + 0x7FFFu + ((u >> 16) & 1u);   // round-to-nearest-even
  return (u16)(r >> 16);
}
__device__ __forceinline__ float bf16_to_f32(u16 h) {
  union { u32 u; float f; } cv; cv.u = ((u32)h) << 16;
  return cv.f;
}
__device__ __forceinline__ v8f v8f_zero() {
  v8f z;
#pragma unroll
  for (int i = 0; i < 8; ++i) z[i] = 0.0f;
  return z;
}

// ---------------------------------------------------------------------------
// 1) fp32 -> bf16 convert (4 elems/thread)
// ---------------------------------------------------------------------------
__global__ __launch_bounds__(256)
void cvt_f32_bf16(const float* __restrict__ in, u16* __restrict__ out, int n4) {
  int i = blockIdx.x * 256 + threadIdx.x;
  if (i < n4) {
    float4 f = ((const float4*)in)[i];
    u32 lo = (u32)f32_to_bf16_rne(f.x) | ((u32)f32_to_bf16_rne(f.y) << 16);
    u32 hi = (u32)f32_to_bf16_rne(f.z) | ((u32)f32_to_bf16_rne(f.w) << 16);
    ((uint2*)out)[i] = make_uint2(lo, hi);
  }
}

// ---------------------------------------------------------------------------
// 2) GEMM: C[M,N] = A[M,K] * B[N,K]^T  (A,B bf16 row-major, K-contiguous)
//    128x128 block, 8 waves (2M x 4N), BK=64, double-buffered async LDS.
// ---------------------------------------------------------------------------
#define BM 128
#define BN 128
#define BK 64
#define LDSW 72   // bf16 row stride: 144 B = 9x16 B, odd dword rotation

template<bool BF16_OUT>
__global__ __launch_bounds__(256)
void gemm_nt(const u16* __restrict__ A, const u16* __restrict__ B,
             void* __restrict__ C, int M, int N, int K) {
  __shared__ u16 sA[2][BM * LDSW];
  __shared__ u16 sB[2][BN * LDSW];
  const int tid  = threadIdx.x;
  const int lane = tid & 31;
  const int wid  = tid >> 5;
  const int wm   = wid >> 2;        // 0..1
  const int wn   = wid & 3;         // 0..3
  const int hf   = lane >> 4;       // lane half (ISA K-interleave select)
  const int l16  = lane & 15;
  const int m0   = blockIdx.y * BM;
  const int n0   = blockIdx.x * BN;

  v8f acc[4][2];
#pragma unroll
  for (int mi = 0; mi < 4; ++mi)
#pragma unroll
    for (int ni = 0; ni < 2; ++ni) acc[mi][ni] = v8f_zero();

  // stage one 128x64 bf16 panel pair into LDS buffer `buf` (8 B128 per thread)
  auto stage = [&](int buf, int k0) {
#pragma unroll
    for (int it = 0; it < 4; ++it) {
      int c   = it * 256 + tid;     // 0..1023
      int row = c >> 3;             // 0..127
      int kc  = (c & 7) << 3;       // 0..56
      cp_b128(&sA[buf][row * LDSW + kc], &A[(size_t)(m0 + row) * K + k0 + kc]);
      cp_b128(&sB[buf][row * LDSW + kc], &B[(size_t)(n0 + row) * K + k0 + kc]);
    }
  };

  stage(0, 0);
  int cur = 0;
  for (int k0 = 0; k0 < K; k0 += BK) {
    const bool nxt = (k0 + BK) < K;
    if (nxt) { stage(cur ^ 1, k0 + BK); wait_async_le8(); }  // prefetch overlaps
    else     { wait_async_0(); }
    __syncthreads();

#pragma unroll
    for (int kk = 0; kk < 2; ++kk) {           // two 16x32 k-steps
      AFrag afr[4], bfr[2];
#pragma unroll
      for (int mi = 0; mi < 4; ++mi) {
        int m = wm * 64 + mi * 16 + l16;
        const u16* p = &sA[cur][m * LDSW + kk * 32 + hf * 8];
        afr[mi].q[0] = *(const uint4*)p;
        afr[mi].q[1] = *(const uint4*)(p + 16);
      }
#pragma unroll
      for (int ni = 0; ni < 2; ++ni) {
        int n = wn * 32 + ni * 16 + l16;
        const u16* p = &sB[cur][n * LDSW + kk * 32 + hf * 8];
        bfr[ni].q[0] = *(const uint4*)p;
        bfr[ni].q[1] = *(const uint4*)(p + 16);
      }
#pragma unroll
      for (int mi = 0; mi < 4; ++mi)
#pragma unroll
        for (int ni = 0; ni < 2; ++ni)
          acc[mi][ni] = __builtin_amdgcn_wmma_f32_16x16x32_bf16(
              false, afr[mi].v, false, bfr[ni].v, (short)0, acc[mi][ni], false, false);
    }
    __syncthreads();                           // all reads of `cur` done
    cur ^= 1;
  }

#pragma unroll
  for (int mi = 0; mi < 4; ++mi)
#pragma unroll
    for (int ni = 0; ni < 2; ++ni)
#pragma unroll
      for (int r = 0; r < 8; ++r) {
        int row = m0 + wm * 64 + mi * 16 + r + 8 * hf;
        int col = n0 + wn * 32 + ni * 16 + l16;
        float v = acc[mi][ni][r];
        if (BF16_OUT) ((u16*)C)[(size_t)row * N + col] = f32_to_bf16_rne(v);
        else          ((float*)C)[(size_t)row * N + col] = v;
      }
}

// ---------------------------------------------------------------------------
// 3) qkv [B,T,3,H,Dh] bf16 -> Q[b,h,t,dh]*0.125, K[b,h,t,dh], V^T[b,h,dh,t]
// ---------------------------------------------------------------------------
__global__ __launch_bounds__(256)
void reorder_qkv(const u16* __restrict__ qkv, u16* __restrict__ Q,
                 u16* __restrict__ Kt, u16* __restrict__ Vt) {
  size_t idx = (size_t)blockIdx.x * 256 + threadIdx.x;  // B*H*T*Dh = 4194304
  int dh = (int)(idx & 63);
  size_t r = idx >> 6;
  int t = (int)(r & 2047); r >>= 11;
  int h = (int)(r & 15);
  int b = (int)(r >> 4);
  size_t base = (size_t)(b * 2048 + t) * 3072 + (size_t)h * 64 + dh;
  u16 qv = qkv[base];
  u16 kv = qkv[base + 1024];
  u16 vv = qkv[base + 2048];
  size_t qi = ((size_t)(b * 16 + h) * 2048 + t) * 64 + dh;
  Q[qi]  = f32_to_bf16_rne(bf16_to_f32(qv) * 0.125f);  // 1/sqrt(Dh)
  Kt[qi] = kv;
  Vt[((size_t)(b * 16 + h) * 64 + dh) * 2048 + t] = vv;
}

// ---------------------------------------------------------------------------
// 4) Flash attention: block = 128 q-rows of one (b,h); 8 waves x 16 rows.
//    K/V tiles staged cooperatively in double-buffered LDS (async prefetch).
// ---------------------------------------------------------------------------
#define KSTR 72    // sK row stride (keys x 64 dh)
#define VSTR 136   // sV / sP row stride (x 128 t-cols)

__global__ __launch_bounds__(256)
void attention_fused(const u16* __restrict__ Q, const u16* __restrict__ Kt,
                     const u16* __restrict__ Vt, u16* __restrict__ att) {
  __shared__ u16 sK[2][128 * KSTR];   // 2 x 18 KB
  __shared__ u16 sV[2][64 * VSTR];    // 2 x 17 KB
  __shared__ u16 sP[8][16 * VSTR];    // 34 KB (per-wave P tiles)
  const int tid = threadIdx.x, lane = tid & 31, wid = tid >> 5;
  const int hf = lane >> 4, l16 = lane & 15;
  const int h = blockIdx.y, b = blockIdx.z;
  const int q0 = blockIdx.x * 128 + wid * 16;
  const size_t bh = (size_t)(b * 16 + h);
  const u16* Qb = Q  + bh * 2048 * 64;
  const u16* Kb = Kt + bh * 2048 * 64;
  const u16* Vb = Vt + bh * 64 * 2048;
  u16* sPw = &sP[wid][0];

  // Q strip 16x64 -> two 16x32 A fragments, resident for the whole loop
  AFrag qf[2];
#pragma unroll
  for (int c = 0; c < 2; ++c) {
    const u16* p = Qb + (size_t)(q0 + l16) * 64 + c * 32 + hf * 8;
    qf[c].q[0] = *(const uint4*)p;
    qf[c].q[1] = *(const uint4*)(p + 16);
  }

  // stage K[kb..kb+128) x 64 and V^T 64 x [kb..kb+128) (8 B128 per thread)
  auto stage = [&](int buf, int kb) {
#pragma unroll
    for (int it = 0; it < 4; ++it) {
      int c = it * 256 + tid;         // 0..1023
      int kr = c >> 3, kc = (c & 7) << 3;      // K: 128 rows x 8 chunks
      cp_b128(&sK[buf][kr * KSTR + kc], Kb + (size_t)(kb + kr) * 64 + kc);
      int vr = c >> 4, tc = (c & 15) << 3;     // V: 64 rows x 16 chunks
      cp_b128(&sV[buf][vr * VSTR + tc], Vb + (size_t)vr * 2048 + kb + tc);
    }
  };

  v8f O[4];
#pragma unroll
  for (int ni = 0; ni < 4; ++ni) O[ni] = v8f_zero();
  float mrow[8], lrow[8];
#pragma unroll
  for (int r = 0; r < 8; ++r) { mrow[r] = -1e30f; lrow[r] = 0.0f; }

  stage(0, 0);
  int cur = 0;
  for (int kb = 0; kb < 2048; kb += 128) {
    const bool nxt = (kb + 128) < 2048;
    if (nxt) { stage(cur ^ 1, kb + 128); wait_async_le8(); }  // prefetch overlaps
    else     { wait_async_0(); }
    __syncthreads();

    // ---- S = Q K^T : 16 WMMAs fed from LDS ----
    v8f S[8];
#pragma unroll
    for (int j = 0; j < 8; ++j) S[j] = v8f_zero();
#pragma unroll
    for (int c = 0; c < 2; ++c)
#pragma unroll
      for (int j = 0; j < 8; ++j) {
        AFrag bf;
        const u16* p = &sK[cur][(j * 16 + l16) * KSTR + c * 32 + hf * 8];
        bf.q[0] = *(const uint4*)p;
        bf.q[1] = *(const uint4*)(p + 16);
        S[j] = __builtin_amdgcn_wmma_f32_16x16x32_bf16(
            false, qf[c].v, false, bf.v, (short)0, S[j], false, false);
      }

    // ---- online softmax; row r lives on one 16-lane half ----
#pragma unroll
    for (int r = 0; r < 8; ++r) {
      float mx = S[0][r];
#pragma unroll
      for (int j = 1; j < 8; ++j) mx = fmaxf(mx, S[j][r]);
#pragma unroll
      for (int off = 8; off >= 1; off >>= 1) mx = fmaxf(mx, __shfl_xor(mx, off, 32));
      float mnew  = fmaxf(mrow[r], mx);
      float scale = __expf(mrow[r] - mnew);
      mrow[r] = mnew;
      lrow[r] *= scale;
#pragma unroll
      for (int ni = 0; ni < 4; ++ni) O[ni][r] *= scale;
      float rs = 0.0f;
#pragma unroll
      for (int j = 0; j < 8; ++j) {
        float p = __expf(S[j][r] - mnew);
        rs += p;
        sPw[(r + 8 * hf) * VSTR + j * 16 + l16] = f32_to_bf16_rne(p);
      }
#pragma unroll
      for (int off = 8; off >= 1; off >>= 1) rs += __shfl_xor(rs, off, 32);
      lrow[r] += rs;
    }

    // ---- O += P V : 16 WMMAs (P from per-wave LDS, V from staged LDS) ----
#pragma unroll
    for (int c = 0; c < 4; ++c) {
      AFrag pf;
      const u16* pp = &sPw[l16 * VSTR + c * 32 + hf * 8];
      pf.q[0] = *(const uint4*)pp;
      pf.q[1] = *(const uint4*)(pp + 16);
#pragma unroll
      for (int ni = 0; ni < 4; ++ni) {
        AFrag bf;
        const u16* vp = &sV[cur][(ni * 16 + l16) * VSTR + c * 32 + hf * 8];
        bf.q[0] = *(const uint4*)vp;
        bf.q[1] = *(const uint4*)(vp + 16);
        O[ni] = __builtin_amdgcn_wmma_f32_16x16x32_bf16(
            false, pf.v, false, bf.v, (short)0, O[ni], false, false);
      }
    }
    __syncthreads();                 // all reads of buffer `cur` done
    cur ^= 1;
  }

  // ---- finalize: att[b, t, h*64+dh] bf16 ----
#pragma unroll
  for (int ni = 0; ni < 4; ++ni)
#pragma unroll
    for (int r = 0; r < 8; ++r) {
      int row = q0 + r + 8 * hf;
      int col = h * 64 + ni * 16 + l16;
      float v = O[ni][r] / lrow[r];
      att[((size_t)(b * 2048) + row) * 1024 + col] = f32_to_bf16_rne(v);
    }
}

// ---------------------------------------------------------------------------
// Launch
// ---------------------------------------------------------------------------
extern "C" void kernel_launch(void* const* d_in, const int* in_sizes, int n_in,
                              void* d_out, int out_size, void* d_ws, size_t ws_size,
                              hipStream_t stream) {
  const float* x    = (const float*)d_in[0];  // [2,2048,1024]
  const float* wqkv = (const float*)d_in[1];  // [3072,1024]
  const float* wout = (const float*)d_in[2];  // [1024,1024]

  char* ws = (char*)d_ws;
  u16* xbf   = (u16*)(ws);                         //  0MB: 4096x1024
  u16* wqbf  = (u16*)(ws + ((size_t)8  << 20));    //  8MB: 3072x1024
  u16* wobf  = (u16*)(ws + ((size_t)14 << 20));    // 14MB: 1024x1024
  u16* qkvbf = (u16*)(ws + ((size_t)16 << 20));    // 16MB: 4096x3072
  u16* Qg    = (u16*)(ws + ((size_t)40 << 20));    // 40MB: [B,H,T,Dh]
  u16* Kg    = (u16*)(ws + ((size_t)48 << 20));    // 48MB: [B,H,T,Dh]
  u16* Vt    = (u16*)(ws + ((size_t)56 << 20));    // 56MB: [B,H,Dh,T]
  u16* att   = (u16*)(ws + ((size_t)64 << 20));    // 64MB: [B*T, 1024]

  // 1) convert inputs to bf16
  cvt_f32_bf16<<<4096, 256, 0, stream>>>(x,    xbf,  4194304 / 4);
  cvt_f32_bf16<<<3072, 256, 0, stream>>>(wqkv, wqbf, 3145728 / 4);
  cvt_f32_bf16<<<1024, 256, 0, stream>>>(wout, wobf, 1048576 / 4);

  // 2) qkv = x @ Wqkv^T  (M=4096, N=3072, K=1024), bf16 out
  gemm_nt<true><<<dim3(3072 / BN, 4096 / BM), 256, 0, stream>>>(
      xbf, wqbf, (void*)qkvbf, 4096, 3072, 1024);

  // 3) split/transpose into Q (pre-scaled), K, V^T
  reorder_qkv<<<16384, 256, 0, stream>>>(qkvbf, Qg, Kg, Vt);

  // 4) fused attention -> att [B*T, 1024] bf16
  attention_fused<<<dim3(16, 16, 2), 256, 0, stream>>>(Qg, Kg, Vt, att);

  // 5) out = att @ Wout^T  (M=4096, N=1024, K=1024), f32 out
  gemm_nt<false><<<dim3(1024 / BN, 4096 / BM), 256, 0, stream>>>(
      att, wobf, d_out, 4096, 1024, 1024);
}
